// NonLocal_37907381354713
// MI455X (gfx1250) — matrix operand order repaired and checked by reading, not compiled, not run on hardware
//
#include <hip/hip_runtime.h>
#include <hip/hip_bf16.h>

#define B_  16
#define C_  1024
#define CI_ 256
#define N_  2048
#define BN_EPS 1e-5f

typedef __bf16 bf16_t;
typedef __attribute__((ext_vector_type(8)))  bf16_t v8bf;
typedef __attribute__((ext_vector_type(16))) bf16_t v16bf;
typedef __attribute__((ext_vector_type(8)))  float  v8f;
typedef __attribute__((ext_vector_type(4)))  float  v4f;

struct bf16x16 { v8bf lo; v8bf hi; };

// A-operand (16x32, MxK): lane l<16 -> row l, K = k0+0..7 / k0+16..23;
// lane l>=16 -> row l-16, K = k0+8..15 / k0+24..31.  (ISA 7.12.2)
__device__ __forceinline__ v16bf load_a16(const bf16_t* base, int row_stride,
                                          int k0, int lane) {
  int r  = lane & 15;
  int kA = k0 + ((lane < 16) ? 0 : 8);
  const bf16_t* p = base + (size_t)r * row_stride + kA;
  bf16x16 t;
  t.lo = *(const v8bf*)(p);
  t.hi = *(const v8bf*)(p + 16);
  return __builtin_bit_cast(v16bf, t);
}

// B-operand (32x16, KxN): lane l -> column l&15, K = k0+8*(l>>4)+0..7 /
// k0+16+8*(l>>4)+0..7.  Memory holds B transposed: [col][k].
__device__ __forceinline__ v16bf load_b16(const bf16_t* base, int col_stride,
                                          int k0, int lane) {
  int c = lane & 15;
  int k = k0 + ((lane >> 4) << 3);
  const bf16_t* p = base + (size_t)c * col_stride + k;
  bf16x16 t;
  t.lo = *(const v8bf*)(p);
  t.hi = *(const v8bf*)(p + 16);
  return __builtin_bit_cast(v16bf, t);
}

__device__ __forceinline__ v8f wmma_bf16(v16bf a, v16bf b, v8f c) {
  return __builtin_amdgcn_wmma_f32_16x16x32_bf16(false, a, false, b,
                                                 (short)0, c, false, false);
}

template <int Ncnt>
__device__ __forceinline__ void wait_asynccnt() {
  asm volatile("s_wait_asynccnt %0" ::"i"(Ncnt) : "memory");
}

// Async-stage one 64row x 32col bf16 block (4 KB) of g into LDS.
// 8 issues/wave x 32 lanes x 16 B = 4096 B.  Tracked by ASYNCcnt.
__device__ __forceinline__ void stage_g(const bf16_t* gQ, bf16_t* dst, int lane) {
#pragma unroll
  for (int i = 0; i < 8; ++i) {
    int off   = (i * 32 + lane) * 16;  // byte offset in 4 KB block
    int row   = off >> 6;              // 64 B per staged row
    int inrow = off & 63;
    const bf16_t* src = gQ + (size_t)row * N_ + (inrow >> 1);
    unsigned int  lds = (unsigned int)(size_t)dst + (unsigned int)off;
    unsigned long long ga = (unsigned long long)(size_t)src;
    asm volatile("global_load_async_to_lds_b128 %0, %1, off"
                 ::"v"(lds), "v"(ga)
                 : "memory");
  }
}

// B-tile read from staged LDS block: row = c*16 + (l&15), 32 elems per row.
__device__ __forceinline__ v16bf load_b_stage(const bf16_t* stg, int c, int lane) {
  int rowi = c * 16 + (lane & 15);
  int kb   = (lane >> 4) << 3;
  const bf16_t* p = stg + (size_t)rowi * 32 + kb;
  bf16x16 t;
  t.lo = *(const v8bf*)(p);
  t.hi = *(const v8bf*)(p + 16);
  return __builtin_bit_cast(v16bf, t);
}

// Build bf16 A-operand from f32 scores in LDS, fusing the softmax 1/sum scale.
__device__ __forceinline__ v16bf make_a_from_S(const float* Sbase, int k0,
                                               int lane, float scale) {
  int r  = lane & 15;
  int kA = k0 + ((lane < 16) ? 0 : 8);
  const float* p = Sbase + (size_t)r * N_ + kA;
  v4f x0 = *(const v4f*)(p);
  v4f x1 = *(const v4f*)(p + 4);
  v4f x2 = *(const v4f*)(p + 16);
  v4f x3 = *(const v4f*)(p + 20);
  bf16x16 t;
#pragma unroll
  for (int i = 0; i < 4; ++i) {
    t.lo[i]     = (bf16_t)(x0[i] * scale);
    t.lo[i + 4] = (bf16_t)(x1[i] * scale);
    t.hi[i]     = (bf16_t)(x2[i] * scale);
    t.hi[i + 4] = (bf16_t)(x3[i] * scale);
  }
  return __builtin_bit_cast(v16bf, t);
}

// ---------------- kernel 0: x [B][C][N] f32 -> xb [B][N][C] bf16 ------------
// LDS-tiled 32x32 transpose: coalesced f32 reads (n-fast) and coalesced bf16
// writes (c-fast).  Padded to 33 columns to avoid LDS bank conflicts.
__global__ __launch_bounds__(256) void cvt_x_kernel(const float* __restrict__ x,
                                                    bf16_t* __restrict__ xb) {
  __shared__ float tile[32][33];
  int b  = blockIdx.z;
  int c0 = blockIdx.y * 32;
  int n0 = blockIdx.x * 32;
  int tn = threadIdx.x & 31;
  int tr = threadIdx.x >> 5;  // 0..7
#pragma unroll
  for (int i = 0; i < 4; ++i) {
    int c = tr + i * 8;
    tile[c][tn] = x[((size_t)b * C_ + c0 + c) * N_ + n0 + tn];
  }
  __syncthreads();
#pragma unroll
  for (int i = 0; i < 4; ++i) {
    int n = tr + i * 8;
    xb[((size_t)b * N_ + n0 + n) * C_ + c0 + tn] = (bf16_t)tile[tn][n];
  }
}

// ---------------- kernel 0b: convert the 4 weight matrices ------------------
__global__ __launch_bounds__(256) void cvt_w_kernel(
    const float* __restrict__ tw, const float* __restrict__ pw,
    const float* __restrict__ gw, const float* __restrict__ ww,
    bf16_t* __restrict__ twb, bf16_t* __restrict__ pwb,
    bf16_t* __restrict__ gwb, bf16_t* __restrict__ wwb) {
  size_t i = (size_t)blockIdx.x * blockDim.x + threadIdx.x;
  twb[i] = (bf16_t)tw[i];
  pwb[i] = (bf16_t)pw[i];
  gwb[i] = (bf16_t)gw[i];
  wwb[i] = (bf16_t)ww[i];
}

// ---------------- kernel 0c: fold BN + bias into scale/shift ----------------
__global__ __launch_bounds__(256) void bn_prep_kernel(
    const float* __restrict__ wb, const float* __restrict__ gamma,
    const float* __restrict__ beta, const float* __restrict__ mean,
    const float* __restrict__ var, float* __restrict__ scl,
    float* __restrict__ shf) {
  int c = blockIdx.x * 256 + threadIdx.x;
  float inv = gamma[c] * rsqrtf(var[c] + BN_EPS);
  scl[c] = inv;
  shf[c] = (wb[c] - mean[c]) * inv + beta[c];
}

// ---------------- kernel 1: projections theta/phi/g -------------------------
// theta,phi -> [B][N][Ci] (contig Ci), g -> [B][Ci][N] (contig N).
// One wave owns a 64-wide o-chunk: 1 A-tile load feeds 12 WMMAs.
__global__ __launch_bounds__(128) void proj_kernel(
    const bf16_t* __restrict__ xb,
    const bf16_t* __restrict__ twb, const bf16_t* __restrict__ pwb,
    const bf16_t* __restrict__ gwb,
    const float* __restrict__ tb, const float* __restrict__ pb,
    const float* __restrict__ gbias,
    bf16_t* __restrict__ th, bf16_t* __restrict__ ph,
    bf16_t* __restrict__ gbuf) {
  int lane = threadIdx.x & 31;
  int wave = threadIdx.x >> 5;
  int n0 = blockIdx.x * 16;
  int b  = blockIdx.y;
  int o0 = wave * 64;
  const bf16_t* abase = xb + ((size_t)b * N_ + n0) * C_;
  v8f dt[4] = {}, dp[4] = {}, dg[4] = {};
  for (int k0 = 0; k0 < C_; k0 += 32) {
    v16bf a = load_a16(abase, C_, k0, lane);
#pragma unroll
    for (int c = 0; c < 4; ++c) {
      v16bf bt = load_b16(twb + (size_t)(o0 + c * 16) * C_, C_, k0, lane);
      dt[c] = wmma_bf16(a, bt, dt[c]);
      v16bf bp = load_b16(pwb + (size_t)(o0 + c * 16) * C_, C_, k0, lane);
      dp[c] = wmma_bf16(a, bp, dp[c]);
      v16bf bg = load_b16(gwb + (size_t)(o0 + c * 16) * C_, C_, k0, lane);
      dg[c] = wmma_bf16(a, bg, dg[c]);
    }
  }
  int col = lane & 15, rb = (lane < 16) ? 0 : 8;
#pragma unroll
  for (int c = 0; c < 4; ++c) {
    int o = o0 + c * 16 + col;
    float btv = tb[o], bpv = pb[o], bgv = gbias[o];
#pragma unroll
    for (int r = 0; r < 8; ++r) {
      int n = n0 + rb + r;
      th[((size_t)b * N_ + n) * CI_ + o]   = (bf16_t)(dt[c][r] + btv);
      ph[((size_t)b * N_ + n) * CI_ + o]   = (bf16_t)(dp[c][r] + bpv);
      gbuf[((size_t)b * CI_ + o) * N_ + n] = (bf16_t)(dg[c][r] + bgv);
    }
  }
}

// ---------------- kernel 2: fused attention ---------------------------------
#define SM_S_BYTES  (32 * N_ * 4)            // 256 KB f32 score row-block
#define SM_RED_OFF  SM_S_BYTES               // 32x4 f32 partials
#define SM_INV_OFF  (SM_RED_OFF + 512)       // 32 f32 row 1/sum
#define SM_STG_OFF  (SM_INV_OFF + 128)       // 4 waves x 2 x 4 KB g staging
#define SM_TOTAL    (SM_STG_OFF + 4 * 8192)
extern __shared__ char smem_raw[];

__global__ __launch_bounds__(128) void attn_kernel(
    const bf16_t* __restrict__ th, const bf16_t* __restrict__ ph,
    const bf16_t* __restrict__ gbuf, bf16_t* __restrict__ yb) {
  float* Sl     = (float*)smem_raw;
  float* red    = (float*)(smem_raw + SM_RED_OFF);
  float* rowinv = (float*)(smem_raw + SM_INV_OFF);
  int tid = threadIdx.x;
  int lane = tid & 31, wave = tid >> 5;
  int n0 = blockIdx.x * 32;
  int b  = blockIdx.y;

  // Phase 1: S[32][2048] = theta(row-block) @ phi^T, into LDS.
  const bf16_t* thbase = th + ((size_t)b * N_ + n0) * CI_;
  v16bf a0[8], a1[8];
#pragma unroll
  for (int kc = 0; kc < 8; ++kc) {
    a0[kc] = load_a16(thbase, CI_, kc * 32, lane);
    a1[kc] = load_a16(thbase + (size_t)16 * CI_, CI_, kc * 32, lane);
  }
  for (int mt = wave; mt < N_ / 16; mt += 4) {
    int m0 = mt * 16;
    const bf16_t* pbase = ph + ((size_t)b * N_ + m0) * CI_;
    __builtin_prefetch(pbase + (size_t)64 * CI_, 0, 1);
    v8f d0 = {}, d1 = {};
    v16bf bcur = load_b16(pbase, CI_, 0, lane);
#pragma unroll
    for (int kc = 0; kc < 8; ++kc) {
      v16bf bnxt;
      if (kc < 7) bnxt = load_b16(pbase, CI_, (kc + 1) * 32, lane);
      d0 = wmma_bf16(a0[kc], bcur, d0);
      d1 = wmma_bf16(a1[kc], bcur, d1);
      bcur = bnxt;
    }
    int col = lane & 15, rb2 = (lane < 16) ? 0 : 8;
#pragma unroll
    for (int r = 0; r < 8; ++r) {
      Sl[(rb2 + r) * N_ + m0 + col]      = d0[r];
      Sl[(16 + rb2 + r) * N_ + m0 + col] = d1[r];
    }
  }
  __syncthreads();

  // Phase 2: per-row max & exp-sum in LDS (4 threads per row).
  int row = tid >> 2, sub = tid & 3;
  float* Sr = Sl + (size_t)row * N_;
  float lmax = -3.0e38f;
  for (int m = sub; m < N_; m += 4) lmax = fmaxf(lmax, Sr[m]);
  red[row * 4 + sub] = lmax;
  __syncthreads();
  float rmax = red[row * 4];
#pragma unroll
  for (int i = 1; i < 4; ++i) rmax = fmaxf(rmax, red[row * 4 + i]);
  float lsum = 0.f;
  for (int m = sub; m < N_; m += 4) {
    float e = __expf(Sr[m] - rmax);
    Sr[m] = e;
    lsum += e;
  }
  __syncthreads();
  red[row * 4 + sub] = lsum;
  __syncthreads();
  if (sub == 0) {
    float s = red[row * 4] + red[row * 4 + 1] + red[row * 4 + 2] + red[row * 4 + 3];
    rowinv[row] = 1.0f / s;
  }
  __syncthreads();

  // Phase 3: y(row-block) = P @ g, P built on the fly from LDS scores.
  // g-tiles async-staged into LDS, double-buffered (ASYNCcnt).
  int ci0 = wave * 64;
  bf16_t* stg0 = (bf16_t*)(smem_raw + SM_STG_OFF) + (size_t)wave * 4096;
  bf16_t* stg1 = stg0 + 2048;
  const bf16_t* gQ = gbuf + ((size_t)b * CI_ + ci0) * N_;
  float sc0 = rowinv[lane & 15];
  float sc1 = rowinv[16 + (lane & 15)];
  v8f od[2][4] = {};
  stage_g(gQ, stg0, lane);
  for (int m0 = 0; m0 < N_; m0 += 32) {
    bf16_t* cur = (m0 & 32) ? stg1 : stg0;
    bf16_t* nxt = (m0 & 32) ? stg0 : stg1;
    if (m0 + 32 < N_) {
      stage_g(gQ + (m0 + 32), nxt, lane);
      wait_asynccnt<8>();   // previous buffer complete (in-order returns)
    } else {
      wait_asynccnt<0>();
    }
    v16bf av0 = make_a_from_S(Sl, m0, lane, sc0);
    v16bf av1 = make_a_from_S(Sl + (size_t)16 * N_, m0, lane, sc1);
#pragma unroll
    for (int c = 0; c < 4; ++c) {
      v16bf bgv = load_b_stage(cur, c, lane);
      od[0][c] = wmma_bf16(av0, bgv, od[0][c]);
      od[1][c] = wmma_bf16(av1, bgv, od[1][c]);
    }
  }
  int col = lane & 15, rb3 = (lane < 16) ? 0 : 8;
#pragma unroll
  for (int t = 0; t < 2; ++t)
#pragma unroll
    for (int c = 0; c < 4; ++c)
#pragma unroll
      for (int r = 0; r < 8; ++r)
        yb[((size_t)b * N_ + n0 + t * 16 + rb3 + r) * CI_ + ci0 + c * 16 + col] =
            (bf16_t)od[t][c][r];
}

// ---------------- kernel 3: z = w_w @ y (+BN fused) + residual --------------
// Wave owns 16 c-tiles: one y B-tile load feeds 16 WMMAs; y read once.
__global__ __launch_bounds__(128) void out_kernel(
    const bf16_t* __restrict__ wwb, const bf16_t* __restrict__ yb,
    const float* __restrict__ scl, const float* __restrict__ shf,
    const float* __restrict__ x, float* __restrict__ out) {
  int lane = threadIdx.x & 31;
  int wave = threadIdx.x >> 5;
  int n0 = blockIdx.x * 16;
  int b  = blockIdx.y;
  int cbase = wave * 256;
  const bf16_t* bbase = yb + ((size_t)b * N_ + n0) * CI_;
  v8f D[16] = {};
  for (int k0 = 0; k0 < CI_; k0 += 32) {
    v16bf bb = load_b16(bbase, CI_, k0, lane);
#pragma unroll
    for (int ct = 0; ct < 16; ++ct) {
      v16bf a = load_a16(wwb + (size_t)(cbase + ct * 16) * CI_, CI_, k0, lane);
      D[ct] = wmma_bf16(a, bb, D[ct]);
    }
  }
  int col = lane & 15, rb = (lane < 16) ? 0 : 8;
  int n = n0 + col;
#pragma unroll
  for (int ct = 0; ct < 16; ++ct)
#pragma unroll
    for (int r = 0; r < 8; ++r) {
      int c = cbase + ct * 16 + rb + r;
      size_t idx = ((size_t)b * C_ + c) * N_ + n;
      out[idx] = D[ct][r] * scl[c] + shf[c] + x[idx];
    }
}

extern "C" void kernel_launch(void* const* d_in, const int* in_sizes, int n_in,
                              void* d_out, int out_size, void* d_ws, size_t ws_size,
                              hipStream_t stream) {
  const float* x       = (const float*)d_in[0];
  const float* theta_w = (const float*)d_in[1];
  const float* theta_b = (const float*)d_in[2];
  const float* phi_w   = (const float*)d_in[3];
  const float* phi_b   = (const float*)d_in[4];
  const float* g_w     = (const float*)d_in[5];
  const float* g_b     = (const float*)d_in[6];
  const float* w_w     = (const float*)d_in[7];
  const float* w_b     = (const float*)d_in[8];
  const float* bn_g    = (const float*)d_in[9];
  const float* bn_b    = (const float*)d_in[10];
  const float* bn_m    = (const float*)d_in[11];
  const float* bn_v    = (const float*)d_in[12];
  float* out = (float*)d_out;

  char* w = (char*)d_ws;
  auto carve = [&](size_t bytes) {
    void* p = (void*)w;
    w += (bytes + 255) & ~(size_t)255;
    return p;
  };
  bf16_t* xb   = (bf16_t*)carve((size_t)B_ * N_ * C_ * 2);
  bf16_t* th   = (bf16_t*)carve((size_t)B_ * N_ * CI_ * 2);
  bf16_t* ph   = (bf16_t*)carve((size_t)B_ * N_ * CI_ * 2);
  bf16_t* gbuf = (bf16_t*)carve((size_t)B_ * CI_ * N_ * 2);
  bf16_t* yb   = (bf16_t*)carve((size_t)B_ * N_ * CI_ * 2);
  bf16_t* twb  = (bf16_t*)carve((size_t)CI_ * C_ * 2);
  bf16_t* pwb  = (bf16_t*)carve((size_t)CI_ * C_ * 2);
  bf16_t* gwb  = (bf16_t*)carve((size_t)CI_ * C_ * 2);
  bf16_t* wwb  = (bf16_t*)carve((size_t)C_ * CI_ * 2);
  float* bnscl = (float*)carve((size_t)C_ * 4);
  float* bnshf = (float*)carve((size_t)C_ * 4);

  cvt_x_kernel<<<dim3(N_ / 32, C_ / 32, B_), 256, 0, stream>>>(x, xb);
  cvt_w_kernel<<<(CI_ * C_) / 256, 256, 0, stream>>>(theta_w, phi_w, g_w, w_w,
                                                     twb, pwb, gwb, wwb);
  bn_prep_kernel<<<C_ / 256, 256, 0, stream>>>(w_b, bn_g, bn_b, bn_m, bn_v,
                                               bnscl, bnshf);
  proj_kernel<<<dim3(N_ / 16, B_), 128, 0, stream>>>(
      xb, twb, pwb, gwb, theta_b, phi_b, g_b, th, ph, gbuf);
  attn_kernel<<<dim3(N_ / 32, B_), 128, SM_TOTAL, stream>>>(th, ph, gbuf, yb);
  out_kernel<<<dim3(N_ / 16, B_), 128, 0, stream>>>(wwb, yb, bnscl, bnshf, x, out);
}
